// PolicyNetwork_79671643341084
// MI455X (gfx1250) — matrix operand order repaired and checked by reading, not compiled, not run on hardware
//
#include <hip/hip_runtime.h>
#include <hip/hip_bf16.h>

// ---- CDNA5 WMMA types ------------------------------------------------------
typedef __attribute__((ext_vector_type(16))) __bf16 v16bf;
typedef __attribute__((ext_vector_type(8)))  float  v8f;

union FragU { v16bf v; uint4 q[2]; };

__device__ __forceinline__ __bf16 f2bf(float f) { return (__bf16)f; }
__device__ __forceinline__ float  lrelu(float x) { return x > 0.f ? x : 0.01f * x; }

// B=512, R=128, A=1024, D=128, AED=256, H1=88 (pad 96), H2=24 (pad 32)
__global__ __launch_bounds__(256)
void policy_mlp_softmax_kernel(const float* __restrict__ reg_emb,
                               const float* __restrict__ W1,
                               const float* __restrict__ b1,
                               const float* __restrict__ W2,
                               const float* __restrict__ b2,
                               const float* __restrict__ W3,
                               const float* __restrict__ nop_table,
                               const int*   __restrict__ action_space,
                               float* __restrict__ out)
{
    // LDS staging (~116 KB of the 320 KB/WGP pool)
    __shared__ __bf16 s_regs[130 * 128];   // reg_emb[b] rows 0..127 + nop rows 128,129
    __shared__ __bf16 s_w1[96 * 256];      // W1 padded rows 88..95 = 0 (row-major, K contiguous)
    __shared__ __bf16 s_w2[32 * 96];       // W2 padded rows 24..31 / cols 88..95 = 0
    __shared__ float  s_b1[96];
    __shared__ float  s_b2[32];
    __shared__ float  s_w3[32];
    __shared__ __bf16 s_h1[8][16 * 96];    // per-wave h1 staging (C-layout -> A-layout)
    __shared__ float  s_logits[1024];
    __shared__ float  s_red[256];

    const int t = threadIdx.x;
    const int b = blockIdx.x;

    // ---- stage + convert to bf16 ------------------------------------------
    const float* remb = reg_emb + (size_t)b * (128 * 128);
    for (int i = t; i < 128 * 128; i += 256) s_regs[i] = f2bf(remb[i]);
    // nop embedding -> leaky_relu -> rows 128,129 (t covers all 256 elems)
    s_regs[16384 + t] = f2bf(lrelu(nop_table[t]));
    for (int i = t; i < 96 * 256; i += 256) {
        const int r = i >> 8;
        s_w1[i] = f2bf(r < 88 ? W1[i] : 0.f);
    }
    for (int i = t; i < 32 * 96; i += 256) {
        const int r = i / 96, c = i - r * 96;
        s_w2[i] = f2bf((r < 24 && c < 88) ? W2[r * 88 + c] : 0.f);
    }
    if (t < 96) s_b1[t] = (t < 88) ? b1[t] : 0.f;
    if (t < 32) s_b2[t] = (t < 24) ? b2[t] : 0.f;
    if (t < 32) s_w3[t] = (t < 24) ? W3[t] : 0.f;
    __syncthreads();

    const int lane = t & 31;
    const int wave = t >> 5;
    const int m    = lane & 15;   // row / column-in-tile for this lane
    const int h    = lane >> 4;   // lane half-select (0/1)
    __bf16* h1w = &s_h1[wave][0];

    // ---- 8 passes: each wave computes 16 action rows per pass --------------
    for (int p = 0; p < 8; ++p) {
        const int arow = p * 128 + wave * 16;
        const int a    = arow + m;
        const int2 act = ((const int2*)action_space)[(size_t)b * 1024 + a];
        const bool isnop = (act.x == 0) && (act.y == 0);
        const int id0 = isnop ? 128 : act.x;
        const int id1 = isnop ? 129 : act.y;

        // A fragments: x = [regs[id0] | regs[id1]], K=256 -> 8 tiles of 32
        // 16-bit A layout: V0-3 = K[h*8 .. h*8+8), V4-7 = K[16+h*8 .. )
        v16bf afrag[8];
        #pragma unroll
        for (int kk = 0; kk < 8; ++kk) {
            const int id = (kk < 4) ? id0 : id1;
            const int kb = (kk & 3) * 32 + h * 8;      // element offset in 128-wide row
            const uint4* pa = (const uint4*)&s_regs[id * 128 + kb];
            FragU f; f.q[0] = pa[0]; f.q[1] = pa[2];   // +16 halves
            afrag[kk] = f.v;
        }

        // ---- layer 1: [16x256] x [256x96] ---------------------------------
        v8f acc[6];
        #pragma unroll
        for (int nt = 0; nt < 6; ++nt) {
            v8f c = {0.f, 0.f, 0.f, 0.f, 0.f, 0.f, 0.f, 0.f};
            const int wr = nt * 16 + m;                // W1 row == output column
            #pragma unroll
            for (int kk = 0; kk < 8; ++kk) {
                // 16-bit B layout: V0-3 = K[h*8 .. h*8+8), V4-7 = K[16+h*8 .. )
                const uint4* pb = (const uint4*)&s_w1[wr * 256 + kk * 32 + h * 8];
                FragU f; f.q[0] = pb[0]; f.q[1] = pb[2];
                c = __builtin_amdgcn_wmma_f32_16x16x32_bf16(
                        false, afrag[kk], false, f.v, (short)0, c, false, false);
            }
            acc[nt] = c;
        }

        // bias + leaky_relu, repack to bf16 A-operand via LDS (transpose)
        #pragma unroll
        for (int nt = 0; nt < 6; ++nt) {
            const int col = nt * 16 + m;
            const float bv = s_b1[col];
            #pragma unroll
            for (int v = 0; v < 8; ++v) {
                const int row = v + h * 8;             // C layout: row = VGPR + half*8
                h1w[row * 96 + col] = f2bf(lrelu(acc[nt][v] + bv));
            }
        }

        // ---- layer 2: [16x96] x [96x32] -----------------------------------
        v16bf a2[3];
        #pragma unroll
        for (int kk = 0; kk < 3; ++kk) {
            const uint4* pa = (const uint4*)&h1w[m * 96 + kk * 32 + h * 8];
            FragU f; f.q[0] = pa[0]; f.q[1] = pa[2];
            a2[kk] = f.v;
        }
        v8f acc2[2];
        #pragma unroll
        for (int nt = 0; nt < 2; ++nt) {
            v8f c = {0.f, 0.f, 0.f, 0.f, 0.f, 0.f, 0.f, 0.f};
            const int wr = nt * 16 + m;
            #pragma unroll
            for (int kk = 0; kk < 3; ++kk) {
                const uint4* pb = (const uint4*)&s_w2[wr * 96 + kk * 32 + h * 8];
                FragU f; f.q[0] = pb[0]; f.q[1] = pb[2];
                c = __builtin_amdgcn_wmma_f32_16x16x32_bf16(
                        false, a2[kk], false, f.v, (short)0, c, false, false);
            }
            acc2[nt] = c;
        }

        // ---- layer 3 + logits: dot over 24 cols, 16-lane reduce -----------
        const float w3a = s_w3[m],     w3b = s_w3[16 + m];   // pads are 0
        const float b2a = s_b2[m],     b2b = s_b2[16 + m];
        #pragma unroll
        for (int v = 0; v < 8; ++v) {
            const float t0 = lrelu(acc2[0][v] + b2a);
            const float t1 = lrelu(acc2[1][v] + b2b);
            float part = t0 * w3a + t1 * w3b;                // pad cols * 0 = 0
            part += __shfl_xor(part, 1, 32);
            part += __shfl_xor(part, 2, 32);
            part += __shfl_xor(part, 4, 32);
            part += __shfl_xor(part, 8, 32);                 // reduce within 16-lane half
            if (m == 0) s_logits[arow + v + h * 8] = part;
        }
    }
    __syncthreads();

    // ---- softmax over the 1024 actions of this batch ----------------------
    float lmax = -3.402823466e+38f;
    for (int i = t; i < 1024; i += 256) lmax = fmaxf(lmax, s_logits[i]);
    s_red[t] = lmax; __syncthreads();
    for (int s = 128; s > 0; s >>= 1) {
        if (t < s) s_red[t] = fmaxf(s_red[t], s_red[t + s]);
        __syncthreads();
    }
    const float gmax = s_red[0]; __syncthreads();
    float lsum = 0.f;
    for (int i = t; i < 1024; i += 256) {
        const float e = __expf(s_logits[i] - gmax);
        s_logits[i] = e;
        lsum += e;
    }
    s_red[t] = lsum; __syncthreads();
    for (int s = 128; s > 0; s >>= 1) {
        if (t < s) s_red[t] += s_red[t + s];
        __syncthreads();
    }
    const float inv = 1.0f / s_red[0];
    float* ob = out + (size_t)b * 1024;
    for (int i = t; i < 1024; i += 256) ob[i] = s_logits[i] * inv;
}

extern "C" void kernel_launch(void* const* d_in, const int* in_sizes, int n_in,
                              void* d_out, int out_size, void* d_ws, size_t ws_size,
                              hipStream_t stream) {
    (void)in_sizes; (void)n_in; (void)out_size; (void)d_ws; (void)ws_size;
    const float* reg_emb   = (const float*)d_in[0];
    const float* W1        = (const float*)d_in[1];
    const float* b1        = (const float*)d_in[2];
    const float* W2        = (const float*)d_in[3];
    const float* b2        = (const float*)d_in[4];
    const float* W3        = (const float*)d_in[5];
    const float* nop_table = (const float*)d_in[6];
    const int*   action    = (const int*)d_in[7];
    float* out = (float*)d_out;

    policy_mlp_softmax_kernel<<<512, 256, 0, stream>>>(
        reg_emb, W1, b1, W2, b2, W3, nop_table, action, out);
}
